// MultiHeadedAttentionBlock_86844238725730
// MI455X (gfx1250) — matrix operand order repaired and checked by reading, not compiled, not run on hardware
//
#include <hip/hip_runtime.h>
#include <hip/hip_bf16.h>

// ---------------- problem constants ----------------
#define B_  8
#define S_  1024
#define D_  256
#define H_  8
#define HD_ 32
#define NEGBIG (-1.0e9f)

typedef _Float16 h8   __attribute__((ext_vector_type(8)));
typedef _Float16 v16h __attribute__((ext_vector_type(16)));
typedef float    v8f  __attribute__((ext_vector_type(8)));
typedef float    fvec4 __attribute__((ext_vector_type(4)));

static __device__ __forceinline__ v16h cat8(h8 lo, h8 hi) {
  return __builtin_shufflevector(lo, hi, 0,1,2,3,4,5,6,7,8,9,10,11,12,13,14,15);
}

// D = A(16x32 f16) * B(32x16 f16) + C(16x16 f32), one wave.
static __device__ __forceinline__ v8f wmma16(v16h a, v16h b, v8f c) {
  return __builtin_amdgcn_wmma_f32_16x16x32_f16(false, a, false, b, (short)0, c, false, false);
}

// A-tile (16 rows x 32 k) from row-major [row][k] storage, row stride ld (elems).
// lane<16: row=lane, K chunks {0..7, 16..23}; lane>=16: row=lane-16, K {8..15, 24..31}.
static __device__ __forceinline__ v16h load_a(const _Float16* p, int ld, int lane) {
  int m = lane & 15, hh = lane >> 4;
  const _Float16* r = p + m * ld;
  return cat8(*(const h8*)(r + hh * 8), *(const h8*)(r + 16 + hh * 8));
}

// B-tile (32 k x 16 n) from "n-major" storage q[n][k] (i.e. B = q^T), row stride ld.
// lane: n = lane&15, K chunks {0..7, 16..23} (lane<16) / {8..15, 24..31} (lane>=16).
static __device__ __forceinline__ v16h load_b(const _Float16* p, int ld, int lane) {
  int n = lane & 15, hh = lane >> 4;
  const _Float16* r = p + n * ld + hh * 8;
  return cat8(*(const h8*)r, *(const h8*)(r + 16));
}

// Per-wave LDS store->load ordering (cross-lane through LDS within one wave).
static __device__ __forceinline__ void wave_lds_fence() {
  __builtin_amdgcn_wave_barrier();
  asm volatile("s_wait_dscnt 0" ::: "memory");
  __builtin_amdgcn_wave_barrier();
}

// Async global->LDS copy of 64 contiguous bytes (4 x b128). INST_OFFSET applies to
// both the memory and LDS addresses, so one base pair covers the whole slice.
static __device__ __forceinline__ void async_copy64(unsigned lds_off, const void* gaddr) {
  asm volatile(
      "global_load_async_to_lds_b128 %0, %1, off\n\t"
      "global_load_async_to_lds_b128 %0, %1, off offset:16\n\t"
      "global_load_async_to_lds_b128 %0, %1, off offset:32\n\t"
      "global_load_async_to_lds_b128 %0, %1, off offset:48"
      :: "v"(lds_off), "v"(gaddr) : "memory");
}
// Retire the oldest 4 async ops (in-order completion), leave next buffer in flight.
static __device__ __forceinline__ void wait_async_le4() {
  asm volatile("s_wait_asynccnt 4" ::: "memory");
}
static __device__ __forceinline__ void wait_async0() {
  asm volatile("s_wait_asynccnt 0" ::: "memory");
}

// ---------------- kernel 1: fp32 -> fp16 convert ----------------
__global__ void cvt_f32_f16(const float* __restrict__ src, _Float16* __restrict__ dst, int n) {
  int i = blockIdx.x * blockDim.x + threadIdx.x;
  if (i < n) dst[i] = (_Float16)src[i];
}

// ---------------- kernel 2: fused QKV projection ----------------
// One wave computes a 16-token x 256-col tile of out = x @ W.T + b.
// TR=false: store row-major [B,H,S,HD]; TR=true: store transposed [B,H,HD,S].
template <bool TR>
static __device__ __forceinline__ void proj_mat(
    const _Float16* __restrict__ X, const _Float16* __restrict__ W,
    const float* __restrict__ bvec, _Float16* __restrict__ dst,
    int b, int s0, int lane) {
  const int colL = lane & 15;
  const int rowAdd = (lane >> 4) * 8;
  v8f acc[16];
#pragma unroll
  for (int n = 0; n < 16; ++n) acc[n] = {};
  for (int kb = 0; kb < D_ / 32; ++kb) {
    v16h a = load_a(X + kb * 32, D_, lane);
    // software-pipelined B tiles: load n+1 while WMMA consumes n
    v16h bt = load_b(W + (size_t)kb * 32, D_, lane);
#pragma unroll
    for (int n = 0; n < 16; ++n) {
      v16h btn;
      if (n < 15)
        btn = load_b(W + (size_t)((n + 1) * 16) * D_ + kb * 32, D_, lane);
      acc[n] = wmma16(a, bt, acc[n]);
      bt = btn;
    }
  }
#pragma unroll
  for (int n = 0; n < 16; ++n) {
    const int col = n * 16 + colL;
    const float bia = bvec[col];
    const int h = col >> 5, hd = col & (HD_ - 1);
#pragma unroll
    for (int r = 0; r < 8; ++r) {
      const float v = acc[n][r] + bia;
      const int srow = s0 + r + rowAdd;
      if (TR) dst[(((size_t)b * H_ + h) * HD_ + hd) * S_ + srow] = (_Float16)v;
      else    dst[(((size_t)b * H_ + h) * S_ + srow) * HD_ + hd] = (_Float16)v;
    }
  }
}

__global__ __launch_bounds__(256) void qkv_proj_kernel(
    const _Float16* __restrict__ Xh,
    const _Float16* __restrict__ Wqh, const _Float16* __restrict__ Wkh,
    const _Float16* __restrict__ Wvh,
    const float* __restrict__ bq, const float* __restrict__ bk, const float* __restrict__ bv,
    _Float16* __restrict__ Qh, _Float16* __restrict__ Kh, _Float16* __restrict__ Vt) {
  const int lane = threadIdx.x & 31;
  const int wv   = threadIdx.x >> 5;
  const int tok0 = blockIdx.x * 128 + wv * 16;   // 16 tokens per wave
  const int b    = tok0 >> 10;
  const int s0   = tok0 & (S_ - 1);
  const _Float16* X = Xh + (size_t)tok0 * D_;

  proj_mat<false>(X, Wqh, bq, Qh, b, s0, lane);
  proj_mat<false>(X, Wkh, bk, Kh, b, s0, lane);
  proj_mat<true >(X, Wvh, bv, Vt, b, s0, lane);
}

// ---------------- kernel 3: fused attention + LayerNorm + output proj ----------------
// workgroup = (b, 16-query tile); 8 waves, wave h handles head h.
__global__ __launch_bounds__(256) void attn_fused_kernel(
    const _Float16* __restrict__ Qh, const _Float16* __restrict__ Kh,
    const _Float16* __restrict__ Vt,
    const unsigned char* __restrict__ kv_mask,   // jnp bool_ -> 1 byte/elem
    const float* __restrict__ bias,              // [B,S,S,H]
    const float* __restrict__ gamma, const float* __restrict__ beta,
    const _Float16* __restrict__ Woh, const float* __restrict__ bo,
    float* __restrict__ out) {
  __shared__ __align__(16) float    sMask[S_];               // 4 KB additive mask
  __shared__ __align__(16) float    sBias[2][16 * 32 * H_];  // 2 x 16 KB [q][k][h], dbl-buf
  __shared__ __align__(16) _Float16 sP[H_][16][32];          // 8 KB per-wave P tiles
  __shared__ __align__(16) float    sO[16 * D_];             // 16 KB attention output
  __shared__ __align__(16) _Float16 sXn[16 * D_];            // 8 KB normalized f16

  const int tid  = threadIdx.x;
  const int lane = tid & 31;
  const int h    = tid >> 5;                   // wave == head
  const int b    = blockIdx.x >> 6;
  const int q0   = (blockIdx.x & 63) * 16;

  for (int k = tid; k < S_; k += 256)
    sMask[k] = kv_mask[b * S_ + k] ? 0.f : NEGBIG;

  const _Float16* Qbase = Qh + (((size_t)b * H_ + h) * S_ + q0) * HD_;
  const v16h qa = load_a(Qbase, HD_, lane);

  v8f o0 = {}, o1 = {};
  float m[8], l[8];
#pragma unroll
  for (int r = 0; r < 8; ++r) { m[r] = -3.0e38f; l[r] = 0.f; }

  const int colL = lane & 15;
  const int rowAdd = (lane >> 4) * 8;
  const float scale = 0.17677669529663687f;    // 1/sqrt(HD)

  // per-thread staging slice: 64 contiguous bytes of [16q x 32k x 8h] f32
  const int qr = tid >> 4, part = tid & 15;
  const unsigned biasOff0 = (unsigned)(uintptr_t)(&sBias[0][qr * 256 + part * 16]);
  const unsigned biasOff1 = (unsigned)(uintptr_t)(&sBias[1][qr * 256 + part * 16]);
  const float* biasRow = bias + ((size_t)(b * S_ + q0 + qr)) * S_ * H_ + part * 16;

  // prime the pipeline: block kb=0 -> buffer 0
  async_copy64(biasOff0, biasRow);

  for (int kb = 0; kb < S_; kb += 32) {
    const int cur = (kb >> 5) & 1;
    __syncthreads();                 // everyone finished reading buffer cur^1
    if (kb + 32 < S_) {
      const float* srcn = biasRow + (size_t)(kb + 32) * H_;
      async_copy64(cur ? biasOff0 : biasOff1, srcn);   // fetch next block
      __builtin_prefetch(srcn + 32 * H_, 0, 1);        // HBM->L2, two blocks ahead
      wait_async_le4();              // retire current buffer's 4 ops only
    } else {
      wait_async0();
    }
    __syncthreads();                 // publish current buffer to all waves
    const float* sB = sBias[cur];

    // scores: one WMMA per 16-key tile (K = HD = 32 in a single op)
    const _Float16* Kb = Kh + (((size_t)b * H_ + h) * S_ + kb) * HD_;
    v8f s0 = {}, s1 = {};
    s0 = wmma16(qa, load_b(Kb, HD_, lane), s0);
    s1 = wmma16(qa, load_b(Kb + 16 * HD_, HD_, lane), s1);

    // online softmax over this 32-key block
#pragma unroll
    for (int r = 0; r < 8; ++r) {
      const int row = r + rowAdd;
      float e0 = s0[r] * scale + sB[row * 256 + colL * 8 + h]        + sMask[kb + colL];
      float e1 = s1[r] * scale + sB[row * 256 + (16 + colL) * 8 + h] + sMask[kb + 16 + colL];
      float t = fmaxf(e0, e1);
      t = fmaxf(t, __shfl_xor(t, 1));
      t = fmaxf(t, __shfl_xor(t, 2));
      t = fmaxf(t, __shfl_xor(t, 4));
      t = fmaxf(t, __shfl_xor(t, 8));          // stays within 16-lane half = this row
      const float mn = fmaxf(m[r], t);
      const float corr = __expf(m[r] - mn);
      m[r] = mn;
      const float p0 = __expf(e0 - mn);
      const float p1 = __expf(e1 - mn);
      float su = p0 + p1;
      su += __shfl_xor(su, 1); su += __shfl_xor(su, 2);
      su += __shfl_xor(su, 4); su += __shfl_xor(su, 8);
      l[r] = l[r] * corr + su;
      o0[r] *= corr; o1[r] *= corr;
      sP[h][row][colL]      = (_Float16)p0;
      sP[h][row][16 + colL] = (_Float16)p1;
    }
    wave_lds_fence();

    // reload P in A-matrix lane pattern from this wave's LDS region
    v16h pa;
    {
      const int mrow = lane & 15, hh = lane >> 4;
      const _Float16* pr = &sP[h][mrow][0];
      pa = cat8(*(const h8*)(pr + hh * 16), *(const h8*)(pr + hh * 16 + 8));
    }
    // V B-tiles from transposed V [B,H,HD,S]: contiguous along keys
    const _Float16* Vb = Vt + (((size_t)b * H_ + h) * HD_) * S_ + kb;
    v16h v0t, v1t;
    {
      const int n = lane & 15, hh = lane >> 4;
      const _Float16* r0 = Vb + (size_t)n * S_ + hh * 16;
      const _Float16* r1 = Vb + (size_t)(16 + n) * S_ + hh * 16;
      v0t = cat8(*(const h8*)r0, *(const h8*)(r0 + 8));
      v1t = cat8(*(const h8*)r1, *(const h8*)(r1 + 8));
    }
    o0 = wmma16(pa, v0t, o0);
    o1 = wmma16(pa, v1t, o1);
  }

  // finalize softmax normalization, scatter heads into [16 x 256] LDS
#pragma unroll
  for (int r = 0; r < 8; ++r) {
    const float inv = 1.f / l[r];
    const int row = r + rowAdd;
    sO[row * D_ + h * HD_ + colL]      = o0[r] * inv;
    sO[row * D_ + h * HD_ + 16 + colL] = o1[r] * inv;
  }
  __syncthreads();

  // LayerNorm: 2 rows per wave, 8 elems per lane
#pragma unroll
  for (int rr = 0; rr < 2; ++rr) {
    const int row = h * 2 + rr;
    const float* px = &sO[row * D_ + lane * 8];
    float x[8];
    fvec4 xa = *(const fvec4*)px;
    fvec4 xb = *(const fvec4*)(px + 4);
#pragma unroll
    for (int i = 0; i < 4; ++i) { x[i] = xa[i]; x[4 + i] = xb[i]; }
    float sum = 0.f, sq = 0.f;
#pragma unroll
    for (int i = 0; i < 8; ++i) { sum += x[i]; sq += x[i] * x[i]; }
    sum += __shfl_xor(sum, 1);  sq += __shfl_xor(sq, 1);
    sum += __shfl_xor(sum, 2);  sq += __shfl_xor(sq, 2);
    sum += __shfl_xor(sum, 4);  sq += __shfl_xor(sq, 4);
    sum += __shfl_xor(sum, 8);  sq += __shfl_xor(sq, 8);
    sum += __shfl_xor(sum, 16); sq += __shfl_xor(sq, 16);
    const float mean = sum * (1.f / D_);
    const float var  = sq * (1.f / D_) - mean * mean;
    const float rstd = rsqrtf(var + 1e-5f);
#pragma unroll
    for (int i = 0; i < 8; ++i) {
      const int c = lane * 8 + i;
      sXn[row * D_ + c] = (_Float16)((x[i] - mean) * rstd * gamma[c] + beta[c]);
    }
  }
  __syncthreads();

  // output projection: wave h computes columns [h*32, h*32+32)
  v8f a0 = {}, a1 = {};
  for (int kb = 0; kb < D_ / 32; ++kb) {
    v16h xa;
    {
      const int mrow = lane & 15, hh = lane >> 4;
      const _Float16* pr = &sXn[mrow * D_ + kb * 32];
      xa = cat8(*(const h8*)(pr + hh * 8), *(const h8*)(pr + 16 + hh * 8));
    }
    v16h w0 = load_b(Woh + (size_t)(h * 32) * D_ + kb * 32, D_, lane);
    v16h w1 = load_b(Woh + (size_t)(h * 32 + 16) * D_ + kb * 32, D_, lane);
    a0 = wmma16(xa, w0, a0);
    a1 = wmma16(xa, w1, a1);
  }
#pragma unroll
  for (int r = 0; r < 8; ++r) {
    const int row = r + rowAdd;
    const int c0 = h * 32 + colL, c1 = h * 32 + 16 + colL;
    const size_t base = ((size_t)b * S_ + q0 + row) * D_;
    out[base + c0] = a0[r] + bo[c0];
    out[base + c1] = a1[r] + bo[c1];
  }
}

// ---------------- host launch ----------------
extern "C" void kernel_launch(void* const* d_in, const int* in_sizes, int n_in,
                              void* d_out, int out_size, void* d_ws, size_t ws_size,
                              hipStream_t stream) {
  (void)in_sizes; (void)n_in; (void)out_size; (void)ws_size;
  const float*         q        = (const float*)d_in[0];
  const unsigned char* kv_mask  = (const unsigned char*)d_in[1];
  const float*         attnbias = (const float*)d_in[2];
  const float* Wq = (const float*)d_in[3];  const float* bq = (const float*)d_in[4];
  const float* Wk = (const float*)d_in[5];  const float* bk = (const float*)d_in[6];
  const float* Wv = (const float*)d_in[7];  const float* bv = (const float*)d_in[8];
  const float* gamma = (const float*)d_in[9];
  const float* beta  = (const float*)d_in[10];
  const float* Wo = (const float*)d_in[11]; const float* bo = (const float*)d_in[12];
  float* out = (float*)d_out;

  // workspace carve-up (f16 buffers), ~16.5 MB total
  char* ws = (char*)d_ws;
  _Float16* Xh  = (_Float16*)ws;  ws += (size_t)B_ * S_ * D_ * sizeof(_Float16);
  _Float16* Wqh = (_Float16*)ws;  ws += (size_t)D_ * D_ * sizeof(_Float16);
  _Float16* Wkh = (_Float16*)ws;  ws += (size_t)D_ * D_ * sizeof(_Float16);
  _Float16* Wvh = (_Float16*)ws;  ws += (size_t)D_ * D_ * sizeof(_Float16);
  _Float16* Woh = (_Float16*)ws;  ws += (size_t)D_ * D_ * sizeof(_Float16);
  _Float16* Qh  = (_Float16*)ws;  ws += (size_t)B_ * H_ * S_ * HD_ * sizeof(_Float16);
  _Float16* Kh  = (_Float16*)ws;  ws += (size_t)B_ * H_ * S_ * HD_ * sizeof(_Float16);
  _Float16* Vt  = (_Float16*)ws;  ws += (size_t)B_ * H_ * S_ * HD_ * sizeof(_Float16);

  const int nq = B_ * S_ * D_;      // 2,097,152
  const int nw = D_ * D_;           // 65,536
  cvt_f32_f16<<<(nq + 255) / 256, 256, 0, stream>>>(q,  Xh,  nq);
  cvt_f32_f16<<<(nw + 255) / 256, 256, 0, stream>>>(Wq, Wqh, nw);
  cvt_f32_f16<<<(nw + 255) / 256, 256, 0, stream>>>(Wk, Wkh, nw);
  cvt_f32_f16<<<(nw + 255) / 256, 256, 0, stream>>>(Wv, Wvh, nw);
  cvt_f32_f16<<<(nw + 255) / 256, 256, 0, stream>>>(Wo, Woh, nw);

  qkv_proj_kernel<<<(B_ * S_) / 128, 256, 0, stream>>>(
      Xh, Wqh, Wkh, Wvh, bq, bk, bv, Qh, Kh, Vt);

  attn_fused_kernel<<<(B_ * S_) / 16, 256, 0, stream>>>(
      Qh, Kh, Vt, kv_mask, attnbias, gamma, beta, Woh, bo, out);
}
